// GINModel_64046552318029
// MI455X (gfx1250) — matrix operand order repaired and checked by reading, not compiled, not run on hardware
//
#include <hip/hip_runtime.h>

#define NN 50000
#define EE 800000
#define GG 512
#define LL 3
#define DD 128
#define LDIM 384  // LL*DD

typedef __attribute__((ext_vector_type(16))) __bf16 v16bf;
typedef __attribute__((ext_vector_type(8)))  __bf16 v8bf;
typedef __attribute__((ext_vector_type(8)))  float  v8f;
typedef __attribute__((ext_vector_type(4)))  unsigned int u32x4;
typedef __attribute__((ext_vector_type(8)))  int          i32x8;
typedef __attribute__((ext_vector_type(4)))  int          i32x4;

union V16 { v16bf v; v8bf h[2]; };

#if defined(__HIP_DEVICE_COMPILE__) && __has_builtin(__builtin_amdgcn_tensor_load_to_lds)
#define HAVE_TDM 1
#else
#define HAVE_TDM 0
#endif

__device__ __forceinline__ __bf16 f2bf(float f) {
  unsigned int u = __builtin_bit_cast(unsigned int, f);
  u += 0x7FFFu + ((u >> 16) & 1u);           // round-to-nearest-even
  unsigned short s = (unsigned short)(u >> 16);
  return __builtin_bit_cast(__bf16, s);
}

#if HAVE_TDM
// Issue one TDM 2D-tile load (bf16 elements) into LDS.
// Descriptor layout per CDNA5 ISA ch.10 §8 (D# groups 0/1; groups 2/3 zero => 2D).
// This toolchain exposes the 6-arg builtin:
//   (uint32x4 g0, int32x8 g1, int32x4, int32x4, int32x8, i32 cpol)
__device__ __forceinline__ void tdm_load_2d(unsigned lds_off, const void* gptr,
                                            unsigned dim0, unsigned dim1,
                                            unsigned tile0, unsigned tile1,
                                            unsigned stride0) {
  unsigned long long ga = (unsigned long long)(uintptr_t)gptr;
  u32x4 g0;
  g0[0] = 1u;                                          // count=1, user descriptor
  g0[1] = lds_off;                                     // lds_addr (bytes)
  g0[2] = (unsigned)(ga & 0xFFFFFFFFu);                // global_addr[31:0]
  g0[3] = (unsigned)((ga >> 32) & 0x01FFFFFFu)         // global_addr[56:32]
        | (2u << 30);                                  // type=2 ("image")
  i32x8 g1;
  g1[0] = 0x00010000;                                  // data_size=1 -> 2-byte elems
  g1[1] = (int)((dim0 & 0xFFFFu) << 16);               // tensor_dim0[15:0]
  g1[2] = (int)((dim0 >> 16) | ((dim1 & 0xFFFFu) << 16)); // dim0[31:16] | dim1[15:0]
  g1[3] = (int)((dim1 >> 16) | (tile0 << 16));         // dim1[31:16] | tile_dim0
  g1[4] = (int)(tile1 & 0xFFFFu);                      // tile_dim1 (tile_dim2=0)
  g1[5] = (int)stride0;                                // tensor_dim0_stride[31:0]
  g1[6] = 0;                                           // stride0[47:32] | dim1_stride lo
  g1[7] = 0;
  i32x4 z4 = {0, 0, 0, 0};
  i32x8 z8 = {0, 0, 0, 0, 0, 0, 0, 0};
  __builtin_amdgcn_tensor_load_to_lds(g0, g1, z4, z4, z8, 0);
}
#endif

__global__ void zero_f32(float* __restrict__ p, int n) {
  int i = blockIdx.x * blockDim.x + threadIdx.x;
  int stride = gridDim.x * blockDim.x;
  for (; i < n; i += stride) p[i] = 0.0f;
}

// One edge handled by 32 threads (4 dims each): gather h[src], atomic-add into agg[dst].
__global__ void edge_scatter(const float* __restrict__ h,
                             const int* __restrict__ src,
                             const int* __restrict__ dst,
                             float* __restrict__ agg) {
  int idx = blockIdx.x * blockDim.x + threadIdx.x;
  if (idx >= EE * 32) return;
  int e = idx >> 5;
  int c = (idx & 31) << 2;
  int s = src[e], d = dst[e];
  const float4 v = *(const float4*)(h + (size_t)s * DD + c);
  float* a = agg + (size_t)d * DD + c;
  atomicAdd(a + 0, v.x);
  atomicAdd(a + 1, v.y);
  atomicAdd(a + 2, v.z);
  atomicAdd(a + 3, v.w);
}

// hsum_bf16 = bf16(hin + agg)   (streaming, HBM/L2-rate)
__global__ void add_convert(const float* __restrict__ hin, const float* __restrict__ agg,
                            __bf16* __restrict__ hsum) {
  int i = blockIdx.x * blockDim.x + threadIdx.x;
  if (i >= NN * DD) return;
  hsum[i] = f2bf(hin[i] + agg[i]);
}

// Pre-transpose+convert all 6 weight matrices: wt[(2l+m)][n][k] = bf16(W[l][k][n])
__global__ void prep_weights(const float* __restrict__ W1, const float* __restrict__ W2,
                             __bf16* __restrict__ wt) {
  int i = blockIdx.x * blockDim.x + threadIdx.x;
  if (i >= 2 * LL * DD * DD) return;
  int m = i / (DD * DD);
  int r = i % (DD * DD);
  int n = r >> 7, k = r & 127;
  int l = m >> 1;
  const float* W = (m & 1) ? (W2 + (size_t)l * DD * DD) : (W1 + (size_t)l * DD * DD);
  wt[(size_t)m * DD * DD + n * DD + k] = f2bf(W[k * DD + n]);
}

// Fused GIN layer MLP on bf16 WMMA; tiles staged by the Tensor Data Mover.
// 256 threads = 8 waves; 128-row tile per block; D = 128.
__global__ __launch_bounds__(256)
void gin_mlp(const __bf16* __restrict__ hsum,
             const __bf16* __restrict__ w1t, const __bf16* __restrict__ w2t,
             const float* __restrict__ b1,
             const float* __restrict__ gamma, const float* __restrict__ beta,
             const float* __restrict__ rm, const float* __restrict__ rv,
             const float* __restrict__ b2,
             const int* __restrict__ batch,
             float* __restrict__ hout, float* __restrict__ pooled, int layer) {
  __shared__ alignas(32) __bf16 lA[DD * DD];   // activation tile (then z tile)
  __shared__ alignas(32) __bf16 lWt[DD * DD];  // W^T (W1 then W2)

  const int tid  = threadIdx.x;
  const int wave = tid >> 5;
  const int lane = tid & 31;
  const int half = lane >> 4;
  const int l15  = lane & 15;
  const int m0   = wave * 16;            // wave's 16-row strip within the tile
  const int mbase = m0 + half * 8;       // C/D-layout row base for this lane
  const int rowBase = blockIdx.x * DD;

#if HAVE_TDM
  if (wave == 0) {
    // A tile: rows [rowBase, rowBase+128), TDM zero-fills rows past NN.
    tdm_load_2d((unsigned)(uintptr_t)(void*)lA, hsum + (size_t)rowBase * DD,
                DD, (unsigned)(NN - rowBase), DD, DD, DD);
    // W1^T tile (dense 128x128 bf16).
    tdm_load_2d((unsigned)(uintptr_t)(void*)lWt, w1t, DD, DD, DD, DD, DD);
    __builtin_amdgcn_s_wait_tensorcnt(0);
  }
  __syncthreads();
#else
  for (int i = tid; i < DD * DD; i += 256) {
    int rG = rowBase + (i >> 7);
    lA[i] = (rG < NN) ? hsum[(size_t)rowBase * DD + i] : f2bf(0.f);
    lWt[i] = w1t[i];
  }
  __syncthreads();
#endif

  // ---------------- GEMM1: z = A @ W1 ----------------
  v8f acc[8] = {};
#pragma unroll
  for (int kk = 0; kk < DD; kk += 32) {
    V16 a;
    const int abase = (m0 + l15) * DD + kk + half * 8;
    a.h[0] = *(const v8bf*)&lA[abase];        // K = kk+c0 .. +7
    a.h[1] = *(const v8bf*)&lA[abase + 16];   // K = kk+c0+16 .. +23
#pragma unroll
    for (int t = 0; t < 8; ++t) {
      V16 b;
      const int bbase = (t * 16 + l15) * DD + kk + half * 8;
      b.h[0] = *(const v8bf*)&lWt[bbase];
      b.h[1] = *(const v8bf*)&lWt[bbase + 16];
      acc[t] = __builtin_amdgcn_wmma_f32_16x16x32_bf16(
          false, a.v, false, b.v, (short)0, acc[t], false, false);
    }
  }

  // Epilogue 1: +b1, BatchNorm(eval), ReLU -> z back into lA (wave-private rows)
#pragma unroll
  for (int t = 0; t < 8; ++t) {
    int n = t * 16 + l15;
    float sc = gamma[n] * rsqrtf(rv[n] + 1e-5f);
    float sh = beta[n] + (b1[n] - rm[n]) * sc;
#pragma unroll
    for (int r = 0; r < 8; ++r) {
      float v = acc[t][r] * sc + sh;
      v = v > 0.f ? v : 0.f;
      lA[(mbase + r) * DD + n] = f2bf(v);
    }
  }
  __syncthreads();  // everyone done reading W1 from lWt

#if HAVE_TDM
  if (wave == 0) {
    tdm_load_2d((unsigned)(uintptr_t)(void*)lWt, w2t, DD, DD, DD, DD, DD);
    __builtin_amdgcn_s_wait_tensorcnt(0);
  }
  __syncthreads();
#else
  for (int i = tid; i < DD * DD; i += 256) lWt[i] = w2t[i];
  __syncthreads();
#endif

  // ---------------- GEMM2: out = z @ W2 ----------------
  v8f acc2[8] = {};
#pragma unroll
  for (int kk = 0; kk < DD; kk += 32) {
    V16 a;
    const int abase = (m0 + l15) * DD + kk + half * 8;
    a.h[0] = *(const v8bf*)&lA[abase];
    a.h[1] = *(const v8bf*)&lA[abase + 16];
#pragma unroll
    for (int t = 0; t < 8; ++t) {
      V16 b;
      const int bbase = (t * 16 + l15) * DD + kk + half * 8;
      b.h[0] = *(const v8bf*)&lWt[bbase];
      b.h[1] = *(const v8bf*)&lWt[bbase + 16];
      acc2[t] = __builtin_amdgcn_wmma_f32_16x16x32_bf16(
          false, a.v, false, b.v, (short)0, acc2[t], false, false);
    }
  }

  // Epilogue 2: +b2, ReLU, store h_out, pool into per-graph sums
#pragma unroll
  for (int r = 0; r < 8; ++r) {
    int rowG = rowBase + mbase + r;
    if (rowG >= NN) continue;
    int g = batch[rowG];
    float* prow = pooled + (size_t)g * LDIM + layer * DD;
    float* orow = hout + (size_t)rowG * DD;
#pragma unroll
    for (int t = 0; t < 8; ++t) {
      int n = t * 16 + l15;
      float v = acc2[t][r] + b2[n];
      v = v > 0.f ? v : 0.f;
      orow[n] = v;
      atomicAdd(&prow[n], v);
    }
  }
}

// Small f32 head: out[m,n] = (relu?) (in[m,:] @ W[:,n] + b[n])
__global__ void linear_head(const float* __restrict__ in, const float* __restrict__ W,
                            const float* __restrict__ b, float* __restrict__ out,
                            int M, int K, int Nout, int relu) {
  int idx = blockIdx.x * blockDim.x + threadIdx.x;
  if (idx >= M * Nout) return;
  int m = idx / Nout, n = idx % Nout;
  const float* ip = in + (size_t)m * K;
  float s = b[n];
  for (int k = 0; k < K; ++k) s += ip[k] * W[(size_t)k * Nout + n];
  if (relu) s = s > 0.f ? s : 0.f;
  out[idx] = s;
}

extern "C" void kernel_launch(void* const* d_in, const int* in_sizes, int n_in,
                              void* d_out, int out_size, void* d_ws, size_t ws_size,
                              hipStream_t stream) {
  const float* x     = (const float*)d_in[0];
  const int*   ei    = (const int*)  d_in[1];
  const int*   batch = (const int*)  d_in[2];
  const float* W1    = (const float*)d_in[3];
  const float* b1    = (const float*)d_in[4];
  const float* gamma = (const float*)d_in[5];
  const float* beta  = (const float*)d_in[6];
  const float* rm    = (const float*)d_in[7];
  const float* rv    = (const float*)d_in[8];
  const float* W2    = (const float*)d_in[9];
  const float* b2    = (const float*)d_in[10];
  const float* l1W   = (const float*)d_in[11];
  const float* l1b   = (const float*)d_in[12];
  const float* l2W   = (const float*)d_in[13];
  const float* l2b   = (const float*)d_in[14];
  float* out = (float*)d_out;

  // workspace layout: hA | hB | agg | pooled | gbuf (f32) | hsum (bf16) | wt (bf16)
  float* ws = (float*)d_ws;
  float* hA     = ws;
  float* hB     = hA + (size_t)NN * DD;
  float* agg    = hB + (size_t)NN * DD;
  float* pooled = agg + (size_t)NN * DD;
  float* gbuf   = pooled + (size_t)GG * LDIM;
  __bf16* hsum  = (__bf16*)(gbuf + (size_t)GG * LDIM);
  __bf16* wt    = hsum + (size_t)NN * DD;   // 6 * DD*DD bf16

  zero_f32<<<256, 256, 0, stream>>>(pooled, GG * LDIM);
  prep_weights<<<(2 * LL * DD * DD + 255) / 256, 256, 0, stream>>>(W1, W2, wt);

  const float* hin = x;
  for (int l = 0; l < LL; ++l) {
    float* hout = (l & 1) ? hB : hA;
    zero_f32<<<2048, 256, 0, stream>>>(agg, NN * DD);
    edge_scatter<<<(EE * 32 + 255) / 256, 256, 0, stream>>>(hin, ei, ei + EE, agg);
    add_convert<<<(NN * DD + 255) / 256, 256, 0, stream>>>(hin, agg, hsum);
    gin_mlp<<<(NN + DD - 1) / DD, 256, 0, stream>>>(
        hsum,
        wt + (size_t)(2 * l) * DD * DD, wt + (size_t)(2 * l + 1) * DD * DD,
        b1 + l * DD, gamma + l * DD, beta + l * DD,
        rm + l * DD, rv + l * DD, b2 + l * DD,
        batch, hout, pooled, l);
    hin = hout;
  }
  linear_head<<<(GG * LDIM + 255) / 256, 256, 0, stream>>>(pooled, l1W, l1b, gbuf, GG, LDIM, LDIM, 1);
  linear_head<<<(GG * 10 + 255) / 256, 256, 0, stream>>>(gbuf, l2W, l2b, out, GG, LDIM, 10, 0);
}
